// ScaleDotProductAttention_36962488549662
// MI455X (gfx1250) — matrix operand order repaired and checked by reading, not compiled, not run on hardware
//
#include <hip/hip_runtime.h>

typedef __attribute__((ext_vector_type(16))) _Float16 v16h;
typedef __attribute__((ext_vector_type(8)))  _Float16 h8;
typedef __attribute__((ext_vector_type(8)))  float    v8f;

constexpr int Bb = 2, Ll = 2048, Hh = 16, Ee = 64;
constexpr int BM = 128;          // query rows per block (8 waves x 16)
constexpr int BN = 32;           // key rows per tile (matches WMMA K=32 for P*V)
constexpr int KSTR = 72;         // K tile LDS row stride (halves), 144B rows, 16B aligned
constexpr int VSTR = 40;         // V^T tile LDS row stride (halves), 80B rows
constexpr int PSTR = 40;         // P scratch row stride (halves)

union AFrag { v16h v; h8 h[2]; _Float16 e[16]; };
union H8U   { h8 v; _Float16 e[8]; };

__device__ __forceinline__ v8f wmma16(v16h a, v16h b, v8f c) {
  return __builtin_amdgcn_wmma_f32_16x16x32_f16(false, a, false, b, (short)0, c,
                                                false, false);
}

// native v_exp_f32 (flush-to-zero is fine: args are <= 0 post max-subtraction)
__device__ __forceinline__ float fexp2(float x) {
  return __builtin_amdgcn_exp2f(x);
}

// 16-lane butterfly max via DPP ROW_XMASK (pure VALU, no DS port)
template<int MASK>
__device__ __forceinline__ float rowxor_max(float x) {
  int t = __builtin_amdgcn_update_dpp(0, __float_as_int(x),
                                      0x160 | MASK, 0xf, 0xf, true);
  return fmaxf(x, __int_as_float(t));
}

// broadcast lane (16*half) value to all lanes of its half: ds_swizzle and_mask=0x10
__device__ __forceinline__ float bcast_half0(float x) {
  return __int_as_float(__builtin_amdgcn_ds_swizzle(__float_as_int(x), 0x0010));
}

// Fill an A-fragment from a global f32 row: cols [0..7]->e[0..7], [16..23]->e[8..15]
__device__ __forceinline__ void qput(AFrag& a, const float* p, float s) {
  float4 f0 = *(const float4*)(p);
  float4 f1 = *(const float4*)(p + 4);
  float4 f2 = *(const float4*)(p + 16);
  float4 f3 = *(const float4*)(p + 20);
  a.e[0]=(_Float16)(f0.x*s);  a.e[1]=(_Float16)(f0.y*s);
  a.e[2]=(_Float16)(f0.z*s);  a.e[3]=(_Float16)(f0.w*s);
  a.e[4]=(_Float16)(f1.x*s);  a.e[5]=(_Float16)(f1.y*s);
  a.e[6]=(_Float16)(f1.z*s);  a.e[7]=(_Float16)(f1.w*s);
  a.e[8]=(_Float16)(f2.x*s);  a.e[9]=(_Float16)(f2.y*s);
  a.e[10]=(_Float16)(f2.z*s); a.e[11]=(_Float16)(f2.w*s);
  a.e[12]=(_Float16)(f3.x*s); a.e[13]=(_Float16)(f3.y*s);
  a.e[14]=(_Float16)(f3.z*s); a.e[15]=(_Float16)(f3.w*s);
}

__global__ __launch_bounds__(256)
void fa_fwd_causal(const float* __restrict__ Q, const float* __restrict__ K,
                   const float* __restrict__ V, const float* __restrict__ scale_p,
                   float* __restrict__ O)
{
  __shared__ __align__(16) _Float16 Ks[BN * KSTR];       // K tile, row-major
  __shared__ __align__(16) _Float16 Vt[Ee * VSTR];       // V tile, transposed [d][n]
  __shared__ __align__(16) _Float16 Pb[8 * 16 * PSTR];   // per-wave P scratch

  const int tid  = threadIdx.x;
  const int lane = tid & 31;
  const int wave = tid >> 5;       // 0..7
  const int nl   = lane & 15;      // lane within half (N / M index)
  const int hl   = lane >> 4;      // half-wave select

  const int bh = blockIdx.y;
  const int b  = bh >> 4;          // H == 16
  const int h  = bh & 15;
  const int r0    = blockIdx.x * BM;
  const int wbase = r0 + wave * 16;     // first query row of this wave

  // fold scale and log2(e) into Q so softmax can use native exp2
  const float qs = scale_p[0] * 1.44269504088896340736f;

  // ---- load this wave's 16x64 Q strip into two A-fragments (E chunks 0..31, 32..63)
  const float* qrow = Q + (((size_t)b * Ll + (wbase + nl)) * Hh + h) * Ee;
  AFrag a0, a1;
  qput(a0, qrow + 8 * hl,      qs);
  qput(a1, qrow + 32 + 8 * hl, qs);

  // ones-column B fragment: column N==0 is 1.0, all other columns 0.
  // P x ones accumulates the softmax denominator in the WMMA pipeline.
  AFrag onesb;
  #pragma unroll
  for (int i = 0; i < 16; ++i) onesb.e[i] = (_Float16)(nl == 0 ? 1.0f : 0.0f);

  v8f   o[4] = {};                 // O accumulator: 16 rows x 64 cols (4 C-frags)
  v8f   o4   = {};                 // row-sum accumulator (denominator), column 0
  float mrow[8];
  #pragma unroll
  for (int r = 0; r < 8; ++r) mrow[r] = -__builtin_inff();

  // running tile pointers: advance by a constant stride, no per-iter recompute
  const int rr = tid >> 3;               // 0..31 key row in tile
  const int cc = (tid & 7) * 8;          // 0..56 column start
  constexpr size_t TSTRIDE = (size_t)BN * Hh * Ee;   // floats per key tile step
  const size_t tile0 = (((size_t)b * Ll + rr) * Hh + h) * Ee + cc;
  const float* kp_g = K + tile0;
  const float* vp_g = V + tile0;

  const int jb_end = (r0 + BM) / BN;
  for (int jb = 0; jb < jb_end; ++jb) {
    // ---- prefetch next tile into near caches (global_prefetch_b8)
    if (jb + 1 < jb_end) {
      __builtin_prefetch(kp_g + TSTRIDE, 0, 3);
      __builtin_prefetch(vp_g + TSTRIDE, 0, 3);
    }
    // ---- cooperative tile load: 256 threads, each 8 floats of K and of V
    {
      float4 k0 = *(const float4*)(kp_g);
      float4 k1 = *(const float4*)(kp_g + 4);
      H8U u;
      u.e[0]=(_Float16)k0.x; u.e[1]=(_Float16)k0.y;
      u.e[2]=(_Float16)k0.z; u.e[3]=(_Float16)k0.w;
      u.e[4]=(_Float16)k1.x; u.e[5]=(_Float16)k1.y;
      u.e[6]=(_Float16)k1.z; u.e[7]=(_Float16)k1.w;
      *(h8*)(&Ks[rr * KSTR + cc]) = u.v;     // aligned ds_store_b128

      float4 v0 = *(const float4*)(vp_g);
      float4 v1 = *(const float4*)(vp_g + 4);
      Vt[(cc + 0) * VSTR + rr] = (_Float16)v0.x;
      Vt[(cc + 1) * VSTR + rr] = (_Float16)v0.y;
      Vt[(cc + 2) * VSTR + rr] = (_Float16)v0.z;
      Vt[(cc + 3) * VSTR + rr] = (_Float16)v0.w;
      Vt[(cc + 4) * VSTR + rr] = (_Float16)v1.x;
      Vt[(cc + 5) * VSTR + rr] = (_Float16)v1.y;
      Vt[(cc + 6) * VSTR + rr] = (_Float16)v1.z;
      Vt[(cc + 7) * VSTR + rr] = (_Float16)v1.w;
    }
    kp_g += TSTRIDE;
    vp_g += TSTRIDE;
    __syncthreads();

    // skip tiles entirely above the diagonal (wave-uniform: EXEC stays all-ones)
    if (jb * BN <= wbase + 15) {
      // ---- S = Q*K^T : two 16x16 C tiles (key cols 0..15, 16..31)
      v8f s[2];
      #pragma unroll
      for (int t = 0; t < 2; ++t) {
        const int n0 = t * 16;
        const _Float16* kp = &Ks[(n0 + nl) * KSTR + 8 * hl];
        AFrag kb0, kb1;
        kb0.h[0] = *(const h8*)(kp);        kb0.h[1] = *(const h8*)(kp + 16);
        kb1.h[0] = *(const h8*)(kp + 32);   kb1.h[1] = *(const h8*)(kp + 48);
        v8f sv = {};
        sv = wmma16(a0.v, kb0.v, sv);
        sv = wmma16(a1.v, kb1.v, sv);
        s[t] = sv;
      }

      // ---- causal mask (elementwise, C layout: row = r+8*hl, col = nl + 16t)
      if (jb * BN + BN - 1 > wbase) {
        #pragma unroll
        for (int r = 0; r < 8; ++r) {
          const int i  = wbase + r + 8 * hl;
          const int j0 = jb * BN + nl;
          if (j0 > i)      s[0][r] = -__builtin_inff();
          if (j0 + 16 > i) s[1][r] = -__builtin_inff();
        }
      }

      // ---- online softmax: row max via DPP butterflies, native v_exp_f32
      #pragma unroll
      for (int r = 0; r < 8; ++r) {
        float t = fmaxf(s[0][r], s[1][r]);
        t = rowxor_max<1>(t);
        t = rowxor_max<2>(t);
        t = rowxor_max<4>(t);
        t = rowxor_max<8>(t);
        const float mnew  = fmaxf(mrow[r], t);
        const float alpha = fexp2(mrow[r] - mnew);
        mrow[r] = mnew;
        s[0][r] = fexp2(s[0][r] - mnew);
        s[1][r] = fexp2(s[1][r] - mnew);
        o[0][r] *= alpha; o[1][r] *= alpha;
        o[2][r] *= alpha; o[3][r] *= alpha;
        o4[r]   *= alpha;             // denominator rescales identically
      }

      // ---- convert P to A-layout through wave-private LDS scratch
      _Float16* pw = &Pb[wave * 16 * PSTR];
      #pragma unroll
      for (int r = 0; r < 8; ++r) {
        pw[(r + 8 * hl) * PSTR + nl]      = (_Float16)s[0][r];
        pw[(r + 8 * hl) * PSTR + 16 + nl] = (_Float16)s[1][r];
      }
      __asm__ volatile("s_wait_dscnt 0" ::: "memory");
      AFrag pa;
      const _Float16* pr = &pw[nl * PSTR + 8 * hl];
      pa.h[0] = *(const h8*)(pr);
      pa.h[1] = *(const h8*)(pr + 16);

      // ---- O += P * V  (B-frags contiguous thanks to transposed V tile)
      #pragma unroll
      for (int f = 0; f < 4; ++f) {
        const _Float16* vp = &Vt[(16 * f + nl) * VSTR + 8 * hl];
        AFrag vb;
        vb.h[0] = *(const h8*)(vp);
        vb.h[1] = *(const h8*)(vp + 16);
        o[f] = wmma16(pa.v, vb.v, o[f]);
      }
      // ---- l += rowsum(P) in the matrix pipeline (replaces shuffle tree)
      o4 = wmma16(pa.v, onesb.v, o4);
    }
    __syncthreads();
  }

  // ---- epilogue: normalize and store [B, L, H, D]
  #pragma unroll
  for (int r = 0; r < 8; ++r) {
    const float lsum = bcast_half0(o4[r]);   // denominator lives in column 0
    const float inv  = 1.0f / lsum;
    const int i = wbase + r + 8 * hl;
    float* orow = O + (((size_t)b * Ll + i) * Hh + h) * Ee;
    orow[nl]      = o[0][r] * inv;
    orow[16 + nl] = o[1][r] * inv;
    orow[32 + nl] = o[2][r] * inv;
    orow[48 + nl] = o[3][r] * inv;
  }
}

extern "C" void kernel_launch(void* const* d_in, const int* in_sizes, int n_in,
                              void* d_out, int out_size, void* d_ws, size_t ws_size,
                              hipStream_t stream) {
  (void)in_sizes; (void)n_in; (void)out_size; (void)d_ws; (void)ws_size;
  const float* Q  = (const float*)d_in[0];
  const float* K  = (const float*)d_in[1];
  const float* V  = (const float*)d_in[2];
  const float* sc = (const float*)d_in[3];
  float* Out = (float*)d_out;
  dim3 grid(Ll / BM, Bb * Hh);
  fa_fwd_causal<<<grid, 256, 0, stream>>>(Q, K, V, sc, Out);
}
